// MultiHeadSelfAttention_10514079941422
// MI455X (gfx1250) — matrix operand order repaired and checked by reading, not compiled, not run on hardware
//
#include <hip/hip_runtime.h>

// ---------------------------------------------------------------------------
// Leverage-masked squared-score attention for MI455X (gfx1250, wave32, WMMA).
// bf16 staged once; all GEMMs are pure b128-load + v_wmma_f32_16x16x32_bf16.
// ---------------------------------------------------------------------------

#define CB   2
#define CS   2048
#define CD   768
#define CH   12
#define CDH  64
#define DAMP 1e-6f

typedef __attribute__((ext_vector_type(16))) __bf16        v16bf;
typedef __attribute__((ext_vector_type(8)))  float         v8f;
typedef __attribute__((ext_vector_type(8)))  unsigned int  v8u;

union BF { v8u u; v16bf b; };

__device__ inline unsigned short f2u16(float f) {
    unsigned u = __builtin_bit_cast(unsigned, f);
    u = (u + 0x7FFFu + ((u >> 16) & 1u)) >> 16;   // round-to-nearest-even
    return (unsigned short)u;
}
__device__ inline unsigned pk2(float lo, float hi) {
    return (unsigned)f2u16(lo) | ((unsigned)f2u16(hi) << 16);
}
__device__ inline float bf2f(unsigned short u) {
    unsigned x = ((unsigned)u) << 16;
    return __builtin_bit_cast(float, x);
}
__device__ inline v8f wmma_bf16(const BF& a, const BF& b, v8f c) {
    return __builtin_amdgcn_wmma_f32_16x16x32_bf16(
        false, a.b, false, b.b, (short)0, c, false, false);
}
#define V8F_ZERO (v8f){0.f,0.f,0.f,0.f,0.f,0.f,0.f,0.f}

// ---------------------------------------------------------------------------
// One-time fp32 -> bf16 staging (8 elements / thread), optional scale fold.
// ---------------------------------------------------------------------------
__global__ __launch_bounds__(256)
void cvt_bf16_kernel(const float* __restrict__ src, unsigned short* __restrict__ dst,
                     float scale, int n8) {
    int i = blockIdx.x * blockDim.x + threadIdx.x;
    if (i >= n8) return;
    const float4* s4 = (const float4*)src;
    float4 a = s4[i * 2], b = s4[i * 2 + 1];
    uint4 o;
    o.x = pk2(a.x * scale, a.y * scale);
    o.y = pk2(a.z * scale, a.w * scale);
    o.z = pk2(b.x * scale, b.y * scale);
    o.w = pk2(b.z * scale, b.w * scale);
    ((uint4*)dst)[i] = o;
}

// ---------------------------------------------------------------------------
// Projection GEMM: out = X @ W^T + bias (bf16 in, bf16 out).
// One wave computes a 32x32 tile: 2 A-frags x 2 B-frags -> 4 WMMA per K=32.
// mode 0: q (pre-scaled 1/8), layout (B,H,S,DH)
// mode 1: k,                  layout (B,H,S,DH)
// mode 2: v,                  layout (B,H,DH,S)  (transposed for attention)
// ---------------------------------------------------------------------------
__global__ __launch_bounds__(128)
void proj_kernel(const unsigned short* __restrict__ Xb,
                 const unsigned short* __restrict__ Wb,
                 const float* __restrict__ bias, unsigned short* __restrict__ out,
                 int mode, float bscale) {
    const int NT2 = CD / 32;                                    // 24 N-supertiles
    int wid  = (blockIdx.x * blockDim.x + threadIdx.x) >> 5;
    int lane = threadIdx.x & 31;
    int mt = wid / NT2, nt = wid % NT2;
    int m0 = mt * 32, n0 = nt * 32;
    int l15 = lane & 15, khalf = lane >> 4;

    const uint4* xr0 = (const uint4*)(Xb + (size_t)(m0 + l15) * CD);
    const uint4* xr1 = (const uint4*)(Xb + (size_t)(m0 + 16 + l15) * CD);
    const uint4* wr0 = (const uint4*)(Wb + (size_t)(n0 + l15) * CD);
    const uint4* wr1 = (const uint4*)(Wb + (size_t)(n0 + 16 + l15) * CD);

    v8f c00 = V8F_ZERO, c01 = V8F_ZERO, c10 = V8F_ZERO, c11 = V8F_ZERO;

    for (int kk = 0; kk < CD; kk += 32) {
        if (kk + 32 < CD) {                                     // global_prefetch_b8
            __builtin_prefetch(xr0 + ((kk + 32) >> 3), 0, 3);
            __builtin_prefetch(xr1 + ((kk + 32) >> 3), 0, 3);
        }
        BF a0, a1, b0, b1;
        #pragma unroll
        for (int g = 0; g < 2; ++g) {                           // K = khalf*16+g*8+2vv
            int ai = (kk + khalf * 16 + g * 8) >> 3;
            uint4 ta = xr0[ai], tb = xr1[ai];
            a0.u[g*4+0]=ta.x; a0.u[g*4+1]=ta.y; a0.u[g*4+2]=ta.z; a0.u[g*4+3]=ta.w;
            a1.u[g*4+0]=tb.x; a1.u[g*4+1]=tb.y; a1.u[g*4+2]=tb.z; a1.u[g*4+3]=tb.w;
        }
        int bi = (kk + khalf * 16) >> 3;                        // K = khalf*16+2v
        uint4 t0 = wr0[bi], t1 = wr0[bi + 1];
        uint4 t2 = wr1[bi], t3 = wr1[bi + 1];
        b0.u[0]=t0.x; b0.u[1]=t0.y; b0.u[2]=t0.z; b0.u[3]=t0.w;
        b0.u[4]=t1.x; b0.u[5]=t1.y; b0.u[6]=t1.z; b0.u[7]=t1.w;
        b1.u[0]=t2.x; b1.u[1]=t2.y; b1.u[2]=t2.z; b1.u[3]=t2.w;
        b1.u[4]=t3.x; b1.u[5]=t3.y; b1.u[6]=t3.z; b1.u[7]=t3.w;
        c00 = wmma_bf16(a0, b0, c00);
        c01 = wmma_bf16(a0, b1, c01);
        c10 = wmma_bf16(a1, b0, c10);
        c11 = wmma_bf16(a1, b1, c11);
    }

    auto store_tile = [&](v8f c, int mbase, int nglob) {
        int hidx = nglob >> 6, dh = nglob & 63;
        float bn = bias[nglob] * bscale;
        if (mode != 2) {
            #pragma unroll
            for (int v = 0; v < 8; ++v) {                       // C: m = v + 8*khalf
                int r = mbase + v + 8 * khalf;
                int bb = r >> 11, s = r & (CS - 1);
                out[(((size_t)bb * CH + hidx) * CS + s) * CDH + dh] = f2u16(c[v] + bn);
            }
        } else {
            #pragma unroll
            for (int v = 0; v < 8; v += 2) {                    // pack 2 rows/store
                int r = mbase + v + 8 * khalf;
                int bb = r >> 11, s = r & (CS - 1);
                unsigned p = pk2(c[v] + bn, c[v + 1] + bn);
                *(unsigned*)(out + (((size_t)bb * CH + hidx) * CDH + dh) * CS + s) = p;
            }
        }
    };
    store_tile(c00, m0,      n0 + l15);
    store_tile(c01, m0,      n0 + 16 + l15);
    store_tile(c10, m0 + 16, n0 + l15);
    store_tile(c11, m0 + 16, n0 + 16 + l15);
}

// ---------------------------------------------------------------------------
// Leverage scores per head (batch 0): G = K^T K + damp*I, Gauss-Jordan invert
// in LDS, then lev[s] = k_s^T inv k_s.  One block per head.
// ---------------------------------------------------------------------------
__global__ __launch_bounds__(256)
void leverage_kernel(const unsigned short* __restrict__ kbf, float* __restrict__ lev) {
    __shared__ float G[64][65];
    __shared__ float Inv[64][65];
    __shared__ float fac[64];
    int h = blockIdx.x, t = threadIdx.x;
    const unsigned short* K0 = kbf + (size_t)h * CS * CDH;      // batch 0

    for (int e = t; e < 64 * 64; e += 256) {
        int i = e >> 6, j = e & 63;
        float acc = (i == j) ? DAMP : 0.f;
        const unsigned short* p = K0;
        for (int s = 0; s < CS; ++s) { acc += bf2f(p[i]) * bf2f(p[j]); p += CDH; }
        G[i][j]   = acc;
        Inv[i][j] = (i == j) ? 1.f : 0.f;
    }
    __syncthreads();

    for (int p = 0; p < 64; ++p) {
        float pinv = 1.0f / G[p][p];
        __syncthreads();
        if (t < 64)        G[p][t]        *= pinv;
        else if (t < 128)  Inv[p][t - 64] *= pinv;
        __syncthreads();
        if (t < 64) fac[t] = (t == p) ? 0.f : G[t][p];
        __syncthreads();
        for (int e = t; e < 64 * 128; e += 256) {
            int r = e >> 7, cc = e & 127;
            if (r != p) {
                float f = fac[r];
                if (cc < 64) G[r][cc]        -= f * G[p][cc];
                else         Inv[r][cc - 64] -= f * Inv[p][cc - 64];
            }
        }
        __syncthreads();
    }

    for (int s = t; s < CS; s += 256) {
        float kr[64];
        #pragma unroll
        for (int i = 0; i < 64; ++i) kr[i] = bf2f(K0[(size_t)s * CDH + i]);
        float acc = 0.f;
        for (int i = 0; i < 64; ++i) {
            float ti = 0.f;
            #pragma unroll 8
            for (int j = 0; j < 64; ++j) ti += Inv[i][j] * kr[j];
            acc += kr[i] * ti;
        }
        lev[(size_t)h * CS + s] = acc;
    }
}

// ---------------------------------------------------------------------------
// Per-head top-k: bitonic sort (ascending) of 2048 scores in LDS, take the
// (S-K)-th value as threshold, emit keep-mask.  One 1024-thread block / head.
// ---------------------------------------------------------------------------
__global__ __launch_bounds__(1024)
void topk_kernel(const float* __restrict__ lev, const int* __restrict__ topk_p,
                 unsigned char* __restrict__ mask) {
    __shared__ float v[CS];
    int h = blockIdx.x, t = threadIdx.x;
    v[t]        = lev[(size_t)h * CS + t];
    v[t + 1024] = lev[(size_t)h * CS + t + 1024];
    __syncthreads();

    for (int ksz = 2; ksz <= CS; ksz <<= 1) {
        for (int j = ksz >> 1; j > 0; j >>= 1) {
            int i   = ((t / j) * (j << 1)) + (t % j);
            int ixj = i + j;
            bool up = ((i & ksz) == 0);
            float a = v[i], b = v[ixj];
            if ((a > b) == up) { v[i] = b; v[ixj] = a; }
            __syncthreads();
        }
    }

    int K = topk_p[0];
    if (K < 1) K = 1; if (K > CS) K = CS;
    float thr = v[CS - K];
    mask[(size_t)h * CS + t]        = (lev[(size_t)h * CS + t]        >= thr) ? 1 : 0;
    mask[(size_t)h * CS + t + 1024] = (lev[(size_t)h * CS + t + 1024] >= thr) ? 1 : 0;
}

// Zero masked k rows (masked columns then get score 0 -> weight 0, so v needs
// no masking: weights already kill its contribution).
__global__ __launch_bounds__(256)
void maskk_kernel(unsigned short* __restrict__ kbf, const unsigned char* __restrict__ mask) {
    int idx = blockIdx.x * blockDim.x + threadIdx.x;            // (b,h,s) flat
    int s = idx & (CS - 1);
    int h = (idx >> 11) % CH;
    if (!mask[(size_t)h * CS + s]) {
        uint4* p = (uint4*)(kbf + (size_t)idx * CDH);
        uint4 z = {0u, 0u, 0u, 0u};
        #pragma unroll
        for (int i = 0; i < 8; ++i) p[i] = z;
    }
}

// ---------------------------------------------------------------------------
// Attention: one wave owns 16 query rows of one (b,h).  Per 32-key block:
//   scores(16x32) = q(16x64) @ k^T        -> 4 WMMA
//   sq = scores^2 ; z += row-sums (via C-frag lanes)
//   sq C-layout -> bf16 A-layout through per-wave LDS (DS in-order per wave)
//   ctx(16x64)  += sq(16x32) @ v(32x64)   -> 4 WMMA (v pre-transposed)
// Finish: z cross-lane reduce (shfl_xor, wave32), ctx /= z, store bf16.
// ---------------------------------------------------------------------------
__global__ __launch_bounds__(128)
void attn_kernel(const unsigned short* __restrict__ qbf,
                 const unsigned short* __restrict__ kbf,
                 const unsigned short* __restrict__ vT,
                 unsigned short* __restrict__ ctx) {
    __shared__ float sq_lds[4][16 * 32];
    int widx = threadIdx.x >> 5, lane = threadIdx.x & 31;
    int wid  = blockIdx.x * 4 + widx;
    int qt = wid & 127;                                         // q tile in [0,128)
    int bh = wid >> 7;                                          // (b*H + h)
    int l15 = lane & 15, khalf = lane >> 4;
    const unsigned short* qb = qbf + (size_t)bh * CS * CDH;
    const unsigned short* kb = kbf + (size_t)bh * CS * CDH;
    const unsigned short* vb = vT  + (size_t)bh * CDH * CS;
    float* L = sq_lds[widx];

    BF qa[2];                                                   // q A-frags (K=0..63)
    {
        const uint4* qr = (const uint4*)(qb + (size_t)(qt * 16 + l15) * CDH);
        #pragma unroll
        for (int f = 0; f < 2; ++f)
            #pragma unroll
            for (int g = 0; g < 2; ++g) {
                uint4 q4 = qr[f * 4 + khalf * 2 + g];
                qa[f].u[g * 4 + 0] = q4.x; qa[f].u[g * 4 + 1] = q4.y;
                qa[f].u[g * 4 + 2] = q4.z; qa[f].u[g * 4 + 3] = q4.w;
            }
    }

    v8f cacc[4];
    #pragma unroll
    for (int i = 0; i < 4; ++i) cacc[i] = V8F_ZERO;
    v8f zacc = V8F_ZERO;

    for (int k0 = 0; k0 < CS; k0 += 32) {
        if (k0 + 32 < CS) {                                     // global_prefetch_b8
            __builtin_prefetch(kb + (size_t)(k0 + 32 + l15) * CDH, 0, 3);
            __builtin_prefetch(vb + (size_t)l15 * CS + k0 + 32, 0, 3);
        }
        #pragma unroll
        for (int tt = 0; tt < 2; ++tt) {                        // two 16-key score tiles
            const uint4* kr = (const uint4*)(kb + (size_t)(k0 + tt * 16 + l15) * CDH);
            BF b0, b1;
            uint4 t0 = kr[khalf * 2],     t1 = kr[khalf * 2 + 1];
            uint4 t2 = kr[4 + khalf * 2], t3 = kr[5 + khalf * 2];
            b0.u[0]=t0.x; b0.u[1]=t0.y; b0.u[2]=t0.z; b0.u[3]=t0.w;
            b0.u[4]=t1.x; b0.u[5]=t1.y; b0.u[6]=t1.z; b0.u[7]=t1.w;
            b1.u[0]=t2.x; b1.u[1]=t2.y; b1.u[2]=t2.z; b1.u[3]=t2.w;
            b1.u[4]=t3.x; b1.u[5]=t3.y; b1.u[6]=t3.z; b1.u[7]=t3.w;
            v8f sc = V8F_ZERO;
            sc = wmma_bf16(qa[0], b0, sc);
            sc = wmma_bf16(qa[1], b1, sc);
            v8f sq = sc * sc;
            zacc += sq;
            #pragma unroll
            for (int v = 0; v < 8; ++v)                         // C(m=v+8*khalf, n=l15)
                L[(v + 8 * khalf) * 32 + tt * 16 + l15] = sq[v];
        }
        BF as;                                                  // sq as bf16 A-frag
        #pragma unroll
        for (int w = 0; w < 8; ++w) {
            int g = w >> 2, vv = w & 3;
            int k = khalf * 16 + g * 8 + vv * 2;
            float2 f2 = *(const float2*)(L + l15 * 32 + k);
            as.u[w] = pk2(f2.x, f2.y);
        }
        #pragma unroll
        for (int nt = 0; nt < 4; ++nt) {                        // ctx += sq @ v
            const uint4* vr = (const uint4*)(vb + (size_t)(nt * 16 + l15) * CS + k0);
            uint4 t0 = vr[khalf * 2], t1 = vr[khalf * 2 + 1];
            BF bv;
            bv.u[0]=t0.x; bv.u[1]=t0.y; bv.u[2]=t0.z; bv.u[3]=t0.w;
            bv.u[4]=t1.x; bv.u[5]=t1.y; bv.u[6]=t1.z; bv.u[7]=t1.w;
            cacc[nt] = wmma_bf16(as, bv, cacc[nt]);
        }
    }

    float zrow[8];
    #pragma unroll
    for (int v = 0; v < 8; ++v) {                               // Σ over 16 N-lanes
        float z = zacc[v];
        z += __shfl_xor(z, 1, 32);
        z += __shfl_xor(z, 2, 32);
        z += __shfl_xor(z, 4, 32);
        z += __shfl_xor(z, 8, 32);
        zrow[v] = z + 1e-12f;
    }

    int b = bh / CH, h = bh % CH;
    #pragma unroll
    for (int nt = 0; nt < 4; ++nt)
        #pragma unroll
        for (int v = 0; v < 8; ++v) {
            float val = cacc[nt][v] / zrow[v];
            int srow = qt * 16 + v + 8 * khalf;
            int col  = h * CDH + nt * 16 + l15;
            ctx[((size_t)b * CS + srow) * CD + col] = f2u16(val);
        }
}

// ---------------------------------------------------------------------------
// Output projection: out = ctx @ Wo^T + bo (bf16 in, fp32 out), 32x32 / wave.
// ---------------------------------------------------------------------------
__global__ __launch_bounds__(128)
void oproj_kernel(const unsigned short* __restrict__ ctx,
                  const unsigned short* __restrict__ Wob,
                  const float* __restrict__ bo, float* __restrict__ out) {
    const int NT2 = CD / 32;
    int wid  = (blockIdx.x * blockDim.x + threadIdx.x) >> 5;
    int lane = threadIdx.x & 31;
    int mt = wid / NT2, nt = wid % NT2;
    int m0 = mt * 32, n0 = nt * 32;
    int l15 = lane & 15, khalf = lane >> 4;

    const uint4* xr0 = (const uint4*)(ctx + (size_t)(m0 + l15) * CD);
    const uint4* xr1 = (const uint4*)(ctx + (size_t)(m0 + 16 + l15) * CD);
    const uint4* wr0 = (const uint4*)(Wob + (size_t)(n0 + l15) * CD);
    const uint4* wr1 = (const uint4*)(Wob + (size_t)(n0 + 16 + l15) * CD);

    v8f c00 = V8F_ZERO, c01 = V8F_ZERO, c10 = V8F_ZERO, c11 = V8F_ZERO;

    for (int kk = 0; kk < CD; kk += 32) {
        if (kk + 32 < CD) {
            __builtin_prefetch(xr0 + ((kk + 32) >> 3), 0, 3);
            __builtin_prefetch(xr1 + ((kk + 32) >> 3), 0, 3);
        }
        BF a0, a1, b0, b1;
        #pragma unroll
        for (int g = 0; g < 2; ++g) {
            int ai = (kk + khalf * 16 + g * 8) >> 3;
            uint4 ta = xr0[ai], tb = xr1[ai];
            a0.u[g*4+0]=ta.x; a0.u[g*4+1]=ta.y; a0.u[g*4+2]=ta.z; a0.u[g*4+3]=ta.w;
            a1.u[g*4+0]=tb.x; a1.u[g*4+1]=tb.y; a1.u[g*4+2]=tb.z; a1.u[g*4+3]=tb.w;
        }
        int bi = (kk + khalf * 16) >> 3;
        uint4 t0 = wr0[bi], t1 = wr0[bi + 1];
        uint4 t2 = wr1[bi], t3 = wr1[bi + 1];
        b0.u[0]=t0.x; b0.u[1]=t0.y; b0.u[2]=t0.z; b0.u[3]=t0.w;
        b0.u[4]=t1.x; b0.u[5]=t1.y; b0.u[6]=t1.z; b0.u[7]=t1.w;
        b1.u[0]=t2.x; b1.u[1]=t2.y; b1.u[2]=t2.z; b1.u[3]=t2.w;
        b1.u[4]=t3.x; b1.u[5]=t3.y; b1.u[6]=t3.z; b1.u[7]=t3.w;
        c00 = wmma_bf16(a0, b0, c00);
        c01 = wmma_bf16(a0, b1, c01);
        c10 = wmma_bf16(a1, b0, c10);
        c11 = wmma_bf16(a1, b1, c11);
    }

    auto store_tile = [&](v8f c, int mbase, int nglob) {
        float bn = bo[nglob];
        #pragma unroll
        for (int v = 0; v < 8; ++v) {
            int r = mbase + v + 8 * khalf;
            out[(size_t)r * CD + nglob] = c[v] + bn;
        }
    };
    store_tile(c00, m0,      n0 + l15);
    store_tile(c01, m0,      n0 + 16 + l15);
    store_tile(c10, m0 + 16, n0 + l15);
    store_tile(c11, m0 + 16, n0 + 16 + l15);
}

// ---------------------------------------------------------------------------
extern "C" void kernel_launch(void* const* d_in, const int* in_sizes, int n_in,
                              void* d_out, int out_size, void* d_ws, size_t ws_size,
                              hipStream_t stream) {
    (void)in_sizes; (void)n_in; (void)out_size; (void)ws_size;
    const float* query = (const float*)d_in[0];
    const float* key   = (const float*)d_in[1];
    const float* value = (const float*)d_in[2];
    const float* Wq = (const float*)d_in[3];
    const float* bq = (const float*)d_in[4];
    const float* Wk = (const float*)d_in[5];
    const float* bk = (const float*)d_in[6];
    const float* Wv = (const float*)d_in[7];
    const float* bv = (const float*)d_in[8];
    const float* Wo = (const float*)d_in[9];
    const float* bo = (const float*)d_in[10];
    const int* topk = (const int*)d_in[11];

    char* ws = (char*)d_ws;
    const size_t hde = (size_t)CB * CH * CS * CDH;              // 3,145,728 elems
    const size_t we  = (size_t)CD * CD;                         //   589,824 elems
    unsigned short* qbf  = (unsigned short*)ws; ws += hde * 2;
    unsigned short* kbf  = (unsigned short*)ws; ws += hde * 2;
    unsigned short* vTb  = (unsigned short*)ws; ws += hde * 2;
    unsigned short* ctxb = (unsigned short*)ws; ws += (size_t)CB * CS * CD * 2;
    unsigned short* Xqb  = (unsigned short*)ws; ws += hde * 2;
    unsigned short* Xkb  = (unsigned short*)ws; ws += hde * 2;
    unsigned short* Xvb  = (unsigned short*)ws; ws += hde * 2;
    unsigned short* Wqb  = (unsigned short*)ws; ws += we * 2;
    unsigned short* Wkb  = (unsigned short*)ws; ws += we * 2;
    unsigned short* Wvb  = (unsigned short*)ws; ws += we * 2;
    unsigned short* Wob  = (unsigned short*)ws; ws += we * 2;
    float*          lev  = (float*)ws;          ws += (size_t)CH * CS * 4;
    unsigned char*  mask = (unsigned char*)ws;

    // Stage bf16 copies (q pre-scaled by 1/sqrt(DH)).
    const int xn8 = (int)(hde / 8), wn8 = (int)(we / 8);
    cvt_bf16_kernel<<<xn8 / 256, 256, 0, stream>>>(query, Xqb, 0.125f, xn8);
    cvt_bf16_kernel<<<xn8 / 256, 256, 0, stream>>>(key,   Xkb, 1.0f,   xn8);
    cvt_bf16_kernel<<<xn8 / 256, 256, 0, stream>>>(value, Xvb, 1.0f,   xn8);
    cvt_bf16_kernel<<<wn8 / 256, 256, 0, stream>>>(Wq, Wqb, 1.0f, wn8);
    cvt_bf16_kernel<<<wn8 / 256, 256, 0, stream>>>(Wk, Wkb, 1.0f, wn8);
    cvt_bf16_kernel<<<wn8 / 256, 256, 0, stream>>>(Wv, Wvb, 1.0f, wn8);
    cvt_bf16_kernel<<<wn8 / 256, 256, 0, stream>>>(Wo, Wob, 1.0f, wn8);

    const int gemm_blocks = (CB * CS / 32) * (CD / 32) / 4;     // 768 (4 waves/blk)
    proj_kernel<<<gemm_blocks, 128, 0, stream>>>(Xqb, Wqb, bq, qbf, 0, 0.125f);
    proj_kernel<<<gemm_blocks, 128, 0, stream>>>(Xkb, Wkb, bk, kbf, 1, 1.0f);
    proj_kernel<<<gemm_blocks, 128, 0, stream>>>(Xvb, Wvb, bv, vTb, 2, 1.0f);
    leverage_kernel<<<CH, 256, 0, stream>>>(kbf, lev);
    topk_kernel<<<CH, 1024, 0, stream>>>(lev, topk, mask);
    maskk_kernel<<<(CB * CH * CS) / 256, 256, 0, stream>>>(kbf, mask);
    attn_kernel<<<(CB * CH * (CS / 16)) / 4, 128, 0, stream>>>(qbf, kbf, vTb, ctxb);
    oproj_kernel<<<gemm_blocks, 128, 0, stream>>>(ctxb, Wob, bo, (float*)d_out);
}
